// MADE_79405355368555
// MI455X (gfx1250) — matrix operand order repaired and checked by reading, not compiled, not run on hardware
//
#include <hip/hip_runtime.h>
#include <hip/hip_bf16.h>
#include <math.h>

typedef _Float16 f16;
typedef __attribute__((ext_vector_type(8)))  _Float16 v8h;
typedef __attribute__((ext_vector_type(16))) _Float16 v16h;
typedef __attribute__((ext_vector_type(8)))  float    v8f;

#define BB 1024
#define NI 32
#define NH 1024

// WGP-scope prefetch (scope 0 = pull into all cache levels; the builtin only
// reaches SCOPE_SE/SYS which skip the WGP cache).
__device__ __forceinline__ void prefetch_wgp(const void* p) {
  asm volatile("global_prefetch_b8 %0, off" :: "v"(p));
}

// ---------------- prep kernels: masked weights + zero init ----------------

__global__ void __launch_bounds__(256) k_zero32(float* __restrict__ p, int n) {
  int i = blockIdx.x * blockDim.x + threadIdx.x;
  if (i < n) p[i] = 0.0f;
}

__global__ void __launch_bounds__(256) k_zero16(f16* __restrict__ p, int n) {
  int i = blockIdx.x * blockDim.x + threadIdx.x;
  if (i < n) p[i] = (f16)0.0f;
}

// M1h[h, j] = f16( W1[h, j] * ((h % 31) >= j) )          (NH x NI, f16)
__global__ void __launch_bounds__(256) k_prep_m1(const float* __restrict__ W1,
                                                 f16* __restrict__ M1h) {
  int idx = blockIdx.x * blockDim.x + threadIdx.x;   // 0 .. NH*NI-1
  int h = idx >> 5, j = idx & 31;
  float v = ((int)(h % 31) >= j) ? W1[idx] : 0.0f;
  M1h[idx] = (f16)v;
}

// M2h[o, k] = f16( W2[o, k] * ((o % 31) >= (k % 31)) )   (NH x NH, f16)
__global__ void __launch_bounds__(256) k_prep_m2(const float* __restrict__ W2,
                                                 f16* __restrict__ M2h) {
  int idx = blockIdx.x * blockDim.x + threadIdx.x;   // 0 .. NH*NH-1
  int o = idx >> 10, k = idx & 1023;
  float v = ((o % 31) >= (k % 31)) ? W2[idx] : 0.0f;
  M2h[idx] = (f16)v;
}

// M3m[o, k] = W3[o, k] * (((o % 32) - 1) >= (k % 31))    (2*NI x NH, fp32)
__global__ void __launch_bounds__(256) k_prep_m3(const float* __restrict__ W3,
                                                 float* __restrict__ M3m) {
  int idx = blockIdx.x * blockDim.x + threadIdx.x;   // 0 .. 64*NH-1
  int o = idx >> 10, k = idx & 1023;
  M3m[idx] = (((int)(o & 31) - 1) >= (int)(k % 31)) ? W3[idx] : 0.0f;
}

// -------- layer 1 (WMMA, K=32, one K-step): h1[m,h] = relu(x @ M1.T + b1) --------
// Roles swapped: A = M1h (rows = hidden h, k-contiguous), B = xh (cols = batch m,
// k-contiguous), D rows = h -> packed b128 stores into row-major h1h[m*NH + h].

__global__ void __launch_bounds__(256) k_layer1(const f16* __restrict__ xh,
                                                const f16* __restrict__ M1h,
                                                const float* __restrict__ b1,
                                                f16* __restrict__ h1h) {
  int tid  = blockIdx.x * blockDim.x + threadIdx.x;
  int wave = tid >> 5;
  int lane = tid & 31;
  int hTile  = wave >> 4;          // 0..31 : 32 hidden units
  int mStrip = wave & 15;          // 0..15 : 64 batch rows
  int hBase = hTile * 32;
  int mBase = mStrip * 64;
  int row = lane & 15;
  int hi  = lane >> 4;

  const f16* A0 = M1h + (hBase + row) * NI;        // k-contiguous weight rows
  const f16* A1 = A0 + 16 * NI;
  int ka = hi * 16;
  v8h a0lo = *(const v8h*)(A0 + ka);
  v8h a0hi = *(const v8h*)(A0 + ka + 8);
  v8h a1lo = *(const v8h*)(A1 + ka);
  v8h a1hi = *(const v8h*)(A1 + ka + 8);
  v16h a0 = __builtin_shufflevector(a0lo, a0hi,
            0, 1, 2, 3, 4, 5, 6, 7, 8, 9, 10, 11, 12, 13, 14, 15);
  v16h a1 = __builtin_shufflevector(a1lo, a1hi,
            0, 1, 2, 3, 4, 5, 6, 7, 8, 9, 10, 11, 12, 13, 14, 15);

  v8f acc[8] = {};
#pragma unroll
  for (int t = 0; t < 4; ++t) {
    int m = mBase + t * 16 + row;
    v16h bf = *(const v16h*)(xh + m * NI + hi * 16);   // k-contiguous batch row
    acc[t]     = __builtin_amdgcn_wmma_f32_16x16x32_f16(
                     false, a0, false, bf, (short)0, acc[t],     false, false);
    acc[4 + t] = __builtin_amdgcn_wmma_f32_16x16x32_f16(
                     false, a1, false, bf, (short)0, acc[4 + t], false, false);
  }

  v8f bias0 = *(const v8f*)(b1 + hBase + 8 * hi);       // 8 consecutive h
  v8f bias1 = *(const v8f*)(b1 + hBase + 16 + 8 * hi);
#pragma unroll
  for (int t = 0; t < 4; ++t) {
    int m = mBase + t * 16 + row;
    v8h p0, p1;
#pragma unroll
    for (int r = 0; r < 8; ++r) {
      float v0 = acc[t][r] + bias0[r];     v0 = v0 > 0.0f ? v0 : 0.0f;
      float v1 = acc[4 + t][r] + bias1[r]; v1 = v1 > 0.0f ? v1 : 0.0f;
      p0[r] = (f16)v0;
      p1[r] = (f16)v1;
    }
    f16* dst = h1h + m * NH + hBase + 8 * hi;           // 16-byte aligned
    *(v8h*)(dst)      = p0;
    *(v8h*)(dst + 16) = p1;
  }
}

// -------- layer 2 (WMMA): h2[m,o] = relu(h1 @ M2.T + b2), all row-major --------
// A = M2h weight rows (o, k-contig), B = h1h batch rows (m, k-contig),
// D rows = o -> packed b128 stores into row-major h2h[m*NH + o].
// One wave: 32(o) x 64(m) tile = 8 v_wmma per K-step, 32 K-steps.

__global__ void __launch_bounds__(256) k_layer2(const f16* __restrict__ h1h,
                                                const f16* __restrict__ M2h,
                                                const float* __restrict__ b2,
                                                f16* __restrict__ h2h) {
  int tid  = blockIdx.x * blockDim.x + threadIdx.x;
  int wave = tid >> 5;
  int lane = tid & 31;
  int oTile  = wave >> 4;          // 0..31 : 32 output features
  int mStrip = wave & 15;          // 0..15 : 64 batch rows
  int oBase = oTile * 32;
  int mBase = mStrip * 64;
  int row = lane & 15;
  int hi  = lane >> 4;

  const f16* A0 = M2h + (oBase + row) * NH;
  const f16* A1 = A0 + 16 * NH;
  const f16* Bp = h1h + (mBase + row) * NH + hi * 16;

  v8f acc[8] = {};

  for (int k0 = 0; k0 < NH; k0 += 32) {
    int ka = k0 + hi * 16;
    v8h a0lo = *(const v8h*)(A0 + ka);
    v8h a0hi = *(const v8h*)(A0 + ka + 8);
    v8h a1lo = *(const v8h*)(A1 + ka);
    v8h a1hi = *(const v8h*)(A1 + ka + 8);
    v16h a0 = __builtin_shufflevector(a0lo, a0hi,
              0, 1, 2, 3, 4, 5, 6, 7, 8, 9, 10, 11, 12, 13, 14, 15);
    v16h a1 = __builtin_shufflevector(a1lo, a1hi,
              0, 1, 2, 3, 4, 5, 6, 7, 8, 9, 10, 11, 12, 13, 14, 15);
    if (k0 + 32 < NH) {
      prefetch_wgp(A0 + ka + 32);                 // weight rows
      prefetch_wgp(A1 + ka + 32);
      prefetch_wgp(Bp + k0 + 32);                 // activation rows
      prefetch_wgp(Bp + 2 * 16 * NH + k0 + 32);
    }
#pragma unroll
    for (int t = 0; t < 4; ++t) {
      v16h bf = *(const v16h*)(Bp + t * 16 * NH + k0);
      acc[t]     = __builtin_amdgcn_wmma_f32_16x16x32_f16(
                       false, a0, false, bf, (short)0, acc[t],     false, false);
      acc[4 + t] = __builtin_amdgcn_wmma_f32_16x16x32_f16(
                       false, a1, false, bf, (short)0, acc[4 + t], false, false);
    }
  }

  v8f bias0 = *(const v8f*)(b2 + oBase + 8 * hi);
  v8f bias1 = *(const v8f*)(b2 + oBase + 16 + 8 * hi);
#pragma unroll
  for (int t = 0; t < 4; ++t) {
    int m = mBase + t * 16 + row;
    v8h p0, p1;
#pragma unroll
    for (int r = 0; r < 8; ++r) {
      float v0 = acc[t][r] + bias0[r];     v0 = v0 > 0.0f ? v0 : 0.0f;
      float v1 = acc[4 + t][r] + bias1[r]; v1 = v1 > 0.0f ? v1 : 0.0f;
      p0[r] = (f16)v0;
      p1[r] = (f16)v1;
    }
    f16* dst = h2h + m * NH + oBase + 8 * hi;     // 16-byte aligned
    *(v8h*)(dst)      = p0;
    *(v8h*)(dst + 16) = p1;
  }
}

// ------ layer 3 + autoregressive update: only the 2 needed output columns ------
// One wave per batch row (1024 waves), coalesced lane reads of row-major h2h,
// wave32 shuffle reduction; lane 0 also maintains the f16 mirror of x.

__global__ void __launch_bounds__(256) k_layer3(const f16* __restrict__ h2h,
                                                const float* __restrict__ M3m,
                                                const float* __restrict__ b3,
                                                const float* __restrict__ inputs,
                                                float* __restrict__ xout,
                                                f16* __restrict__ xh,
                                                float* __restrict__ J, int i) {
  int tid  = blockIdx.x * blockDim.x + threadIdx.x;
  int brow = tid >> 5;             // batch row
  int lane = tid & 31;
  const f16*  hrow = h2h + brow * NH;
  const float* w_m = M3m + i * NH;
  const float* w_a = M3m + (NI + i) * NH;
  float sm = 0.0f, sa = 0.0f;
#pragma unroll 8
  for (int t = 0; t < NH / 32; ++t) {
    int k = lane + t * 32;
    float hv = (float)hrow[k];
    sm = fmaf(hv, w_m[k], sm);
    sa = fmaf(hv, w_a[k], sa);
  }
#pragma unroll
  for (int off = 16; off > 0; off >>= 1) {
    sm += __shfl_xor(sm, off, 32);
    sa += __shfl_xor(sa, off, 32);
  }
  if (lane == 0) {
    float m_i = sm + b3[i];
    float a_i = tanhf(sa + b3[NI + i]);
    float x_i = inputs[brow * NI + i] * expf(a_i) + m_i;
    xout[brow * NI + i] = x_i;
    xh[brow * NI + i]   = (f16)x_i;    // f16 mirror feeds layer-1 WMMA
    J[brow] -= a_i;
  }
}

// ---------------------------------- launch ----------------------------------

extern "C" void kernel_launch(void* const* d_in, const int* in_sizes, int n_in,
                              void* d_out, int out_size, void* d_ws, size_t ws_size,
                              hipStream_t stream) {
  const float* inputs = (const float*)d_in[0];
  const float* W1 = (const float*)d_in[1];
  const float* b1 = (const float*)d_in[2];
  const float* W2 = (const float*)d_in[3];
  const float* b2 = (const float*)d_in[4];
  const float* W3 = (const float*)d_in[5];
  const float* b3 = (const float*)d_in[6];

  char* ws = (char*)d_ws;
  float* M3m = (float*)(ws + 0);            // 64*NH*4  = 256 KB
  f16*   M1h = (f16*)  (ws + (256 << 10));  // NH*NI*2  =  64 KB
  f16*   xh  = (f16*)  (ws + (320 << 10));  // BB*NI*2  =  64 KB
  f16*   M2h = (f16*)  (ws + (384 << 10));  // NH*NH*2  =   2 MB
  f16*   h1h = (f16*)  (ws + (384 << 10) + (2 << 20));   // row-major (BB x NH)
  f16*   h2h = (f16*)  (ws + (384 << 10) + (4 << 20));   // row-major (BB x NH)

  float* xout = (float*)d_out;              // (BB, NI) — first output of tuple
  float* J    = xout + BB * NI;             // (BB, 1)  — second output

  // Deterministic per-call prep (masks derived from indices; x/xh/J zeroed).
  k_zero32 <<<(BB * NI + BB + 255) / 256, 256, 0, stream>>>(xout, BB * NI + BB);
  k_zero16 <<<(BB * NI) / 256,        256, 0, stream>>>(xh, BB * NI);
  k_prep_m1<<<(NH * NI) / 256,        256, 0, stream>>>(W1, M1h);
  k_prep_m2<<<(NH * NH) / 256,        256, 0, stream>>>(W2, M2h);
  k_prep_m3<<<(2 * NI * NH) / 256,    256, 0, stream>>>(W3, M3m);

  // 32 serially-dependent autoregressive steps — all three layers on WMMA-or-
  // coalesced paths; only x[:,i]/J epilogue math is scalar fp32.
  for (int i = 0; i < NI; ++i) {
    k_layer1<<<64,  256, 0, stream>>>(xh, M1h, b1, h1h);    // 512 waves, 8 wmma each
    k_layer2<<<64,  256, 0, stream>>>(h1h, M2h, b2, h2h);   // 512 waves, 256 wmma each
    k_layer3<<<128, 256, 0, stream>>>(h2h, M3m, b3, inputs, xout, xh, J, i);
  }
}
